// MultiHeadAttention_18159121728176
// MI455X (gfx1250) — compile-verified, matrix operand order checked by reading, not run on hardware
//
#include <hip/hip_runtime.h>
#include <hip/hip_fp16.h>

typedef __attribute__((ext_vector_type(16))) _Float16 v16h;
typedef __attribute__((ext_vector_type(8)))  float    v8f;

#define D_MODEL 768
#define NH      12
#define DH      64
#define BB      4
#define LL      2048
#define ML      (BB*LL)       /* 8192 rows */
#define E3      (3*D_MODEL)   /* 2304 */

union AFrag { v16h h; unsigned u[8]; };

__device__ __forceinline__ unsigned ld_u32g(const _Float16* p) {
  return *reinterpret_cast<const unsigned*>(p);
}

// K-index of the low half of A/B-fragment VGPR v for this lane (16-bit 16x32 layout, ISA 7.12.2)
__device__ __forceinline__ int kmap(int lane, int v) {
  return ((lane >> 4) & 1) * 8 + ((v < 4) ? (2 * v) : (16 + 2 * (v - 4)));
}

__device__ __forceinline__ v8f wmma_f16(const AFrag& a, const AFrag& b, v8f c) {
  return __builtin_amdgcn_wmma_f32_16x16x32_f16(false, a.h, false, b.h, (short)0, c, false, false);
}

// one 16x64 GEMM tile's worth of fragments for one 32-wide K chunk
struct Frags { AFrag a, b0, b1, b2, b3; };

__device__ __forceinline__ void load_frags(Frags& f,
                                           const _Float16* arow,
                                           const _Float16* b0r, const _Float16* b1r,
                                           const _Float16* b2r, const _Float16* b3r,
                                           int k0, const int* km) {
#pragma unroll
  for (int v = 0; v < 8; ++v) {
    f.a.u[v]  = ld_u32g(arow + k0 + km[v]);
    f.b0.u[v] = ld_u32g(b0r  + k0 + km[v]);
    f.b1.u[v] = ld_u32g(b1r  + k0 + km[v]);
    f.b2.u[v] = ld_u32g(b2r  + k0 + km[v]);
    f.b3.u[v] = ld_u32g(b3r  + k0 + km[v]);
  }
}

// ---------------------------------------------------------------- converts
__global__ void f32_to_f16_kernel(const float* __restrict__ in, _Float16* __restrict__ out, int n) {
  int i = blockIdx.x * blockDim.x + threadIdx.x;
  int stride = gridDim.x * blockDim.x;
  for (; i < n; i += stride) out[i] = (_Float16)in[i];
}

// ---------------------------------------------------------------- QKV projection
// qkv[r,e] = sum_k xh[r,k]*wh[e,k] + b[e]; scatter into Q,K [b][h][l][64] and V^T [b][h][64][L]
__global__ void qkv_gemm_kernel(const _Float16* __restrict__ xh,
                                const _Float16* __restrict__ wh,
                                const float*    __restrict__ bqkv,
                                _Float16* __restrict__ Qh,
                                _Float16* __restrict__ Kh,
                                _Float16* __restrict__ Vt) {
  const int lane = threadIdx.x & 31;
  const int wid  = blockIdx.x * (blockDim.x >> 5) + (threadIdx.x >> 5);
  const int ntiles = E3 / 64;                 // 36
  const int tm = wid / ntiles;                // 0..511
  const int tn = wid % ntiles;
  const int ln = lane & 15;
  const int hi = lane >> 4;

  int km[8];
#pragma unroll
  for (int v = 0; v < 8; ++v) km[v] = kmap(lane, v);

  const _Float16* arow = xh + (tm * 16 + ln) * D_MODEL;
  const _Float16* b0r  = wh + (tn * 64 +  0 + ln) * D_MODEL;
  const _Float16* b1r  = wh + (tn * 64 + 16 + ln) * D_MODEL;
  const _Float16* b2r  = wh + (tn * 64 + 32 + ln) * D_MODEL;
  const _Float16* b3r  = wh + (tn * 64 + 48 + ln) * D_MODEL;

  v8f acc0 = {}, acc1 = {}, acc2 = {}, acc3 = {};
  Frags fA, fB;
  load_frags(fA, arow, b0r, b1r, b2r, b3r, 0, km);
  for (int k0 = 0; k0 < D_MODEL; k0 += 64) {
    load_frags(fB, arow, b0r, b1r, b2r, b3r, k0 + 32, km);  // in flight over set-A WMMAs
    acc0 = wmma_f16(fA.a, fA.b0, acc0);
    acc1 = wmma_f16(fA.a, fA.b1, acc1);
    acc2 = wmma_f16(fA.a, fA.b2, acc2);
    acc3 = wmma_f16(fA.a, fA.b3, acc3);
    if (k0 + 64 < D_MODEL)
      load_frags(fA, arow, b0r, b1r, b2r, b3r, k0 + 64, km); // in flight over set-B WMMAs
    acc0 = wmma_f16(fB.a, fB.b0, acc0);
    acc1 = wmma_f16(fB.a, fB.b1, acc1);
    acc2 = wmma_f16(fB.a, fB.b2, acc2);
    acc3 = wmma_f16(fB.a, fB.b3, acc3);
  }

  v8f accs[4] = {acc0, acc1, acc2, acc3};
#pragma unroll
  for (int g = 0; g < 4; ++g) {
#pragma unroll
    for (int v = 0; v < 8; ++v) {
      int r = tm * 16 + v + 8 * hi;
      int e = tn * 64 + g * 16 + ln;
      float val = accs[g][v] + bqkv[e];
      int b   = r / LL, l = r % LL;
      int s   = e / D_MODEL;
      int rem = e % D_MODEL;
      int h   = rem / DH, d = rem % DH;
      _Float16 hv = (_Float16)val;
      if (s == 0)      Qh[(((size_t)b * NH + h) * LL + l) * DH + d] = hv;
      else if (s == 1) Kh[(((size_t)b * NH + h) * LL + l) * DH + d] = hv;
      else             Vt[(((size_t)b * NH + h) * DH + d) * LL + l] = hv;
    }
  }
}

// ---------------------------------------------------------------- flash attention
// one wave per (b,h,16-row q tile); streams 32 keys/iter; online softmax in C-fragment layout
__global__ void attn_kernel(const _Float16* __restrict__ Qh,
                            const _Float16* __restrict__ Kh,
                            const _Float16* __restrict__ Vt,
                            _Float16* __restrict__ AO) {
  __shared__ _Float16 pbuf[4][16 * 36];   // per-wave 16x32 P tile, padded stride 36 halves
  const int lane = threadIdx.x & 31;
  const int w    = threadIdx.x >> 5;
  const int wid  = blockIdx.x * (blockDim.x >> 5) + w;
  const int qtiles = LL / 16;             // 128
  const int bh = wid / qtiles;            // 0..47
  const int qt = wid % qtiles;
  const int ln = lane & 15;
  const int hi = lane >> 4;

  const _Float16* Qb = Qh + (size_t)bh * LL * DH;
  const _Float16* Kb = Kh + (size_t)bh * LL * DH;
  const _Float16* Vb = Vt + (size_t)bh * DH * LL;

  int km[8];
#pragma unroll
  for (int v = 0; v < 8; ++v) km[v] = kmap(lane, v);

  // Q A-fragments (d = 0..31 and 32..63)
  const _Float16* qrow = Qb + (qt * 16 + ln) * DH;
  AFrag qa0, qa1;
#pragma unroll
  for (int v = 0; v < 8; ++v) {
    qa0.u[v] = ld_u32g(qrow + km[v]);
    qa1.u[v] = ld_u32g(qrow + 32 + km[v]);
  }

  float mrun[8], lrun[8];
#pragma unroll
  for (int v = 0; v < 8; ++v) { mrun[v] = -1e30f; lrun[v] = 0.0f; }
  v8f o0 = {}, o1 = {}, o2 = {}, o3 = {};

  _Float16* myp = pbuf[w];
  const float scale = 0.125f;             // 1/sqrt(64)
  const int nblk = (qt + 2) >> 1;         // causal: ceil(16*(qt+1)/32)

  for (int jb = 0; jb < nblk; ++jb) {
    const int key0 = jb * 32;
    const _Float16* krow0 = Kb + (key0 + ln) * DH;
    const _Float16* krow1 = Kb + (key0 + 16 + ln) * DH;
    // K fragments for this block
    AFrag kf0, kf1, kf2, kf3;
#pragma unroll
    for (int v = 0; v < 8; ++v) {
      kf0.u[v] = ld_u32g(krow0 + km[v]);
      kf1.u[v] = ld_u32g(krow0 + 32 + km[v]);
      kf2.u[v] = ld_u32g(krow1 + km[v]);
      kf3.u[v] = ld_u32g(krow1 + 32 + km[v]);
    }
    // V^T fragments: independent of softmax -> issue now, overlap S WMMAs + VALU
    const _Float16* vr0 = Vb + ( 0 + ln) * LL + key0;
    const _Float16* vr1 = Vb + (16 + ln) * LL + key0;
    const _Float16* vr2 = Vb + (32 + ln) * LL + key0;
    const _Float16* vr3 = Vb + (48 + ln) * LL + key0;
    AFrag vf0, vf1, vf2, vf3;
#pragma unroll
    for (int v = 0; v < 8; ++v) {
      vf0.u[v] = ld_u32g(vr0 + km[v]);
      vf1.u[v] = ld_u32g(vr1 + km[v]);
      vf2.u[v] = ld_u32g(vr2 + km[v]);
      vf3.u[v] = ld_u32g(vr3 + km[v]);
    }
    // warm WGP$ for next block's K rows
    if (jb + 1 < nblk) {
      __builtin_prefetch(krow0 + 32 * DH, 0, 3);
      __builtin_prefetch(krow1 + 32 * DH, 0, 3);
    }

    v8f s0 = {}, s1 = {};
    s0 = wmma_f16(qa0, kf0, s0);
    s0 = wmma_f16(qa1, kf1, s0);
    s1 = wmma_f16(qa0, kf2, s1);
    s1 = wmma_f16(qa1, kf3, s1);

#pragma unroll
    for (int v = 0; v < 8; ++v) {
      const int q = qt * 16 + v + 8 * hi;
      float x0 = s0[v] * scale;
      float x1 = s1[v] * scale;
      if (key0 + ln > q)      x0 = -1e30f;
      if (key0 + 16 + ln > q) x1 = -1e30f;
      float mx = fmaxf(x0, x1);
#pragma unroll
      for (int d = 1; d < 16; d <<= 1) mx = fmaxf(mx, __shfl_xor(mx, d, 32));
      const float mn   = fmaxf(mrun[v], mx);
      const float corr = __expf(mrun[v] - mn);
      const float p0   = __expf(x0 - mn);
      const float p1   = __expf(x1 - mn);
      float rs = p0 + p1;
#pragma unroll
      for (int d = 1; d < 16; d <<= 1) rs += __shfl_xor(rs, d, 32);
      lrun[v] = lrun[v] * corr + rs;
      mrun[v] = mn;
      o0[v] *= corr; o1[v] *= corr; o2[v] *= corr; o3[v] *= corr;
      const int m = v + 8 * hi;
      myp[m * 36 + ln]      = (_Float16)p0;
      myp[m * 36 + 16 + ln] = (_Float16)p1;
    }
    asm volatile("s_wait_dscnt 0" ::: "memory");   // order wave-local LDS store -> load

    AFrag pa;
#pragma unroll
    for (int v = 0; v < 8; ++v)
      pa.u[v] = *reinterpret_cast<const unsigned*>(&myp[ln * 36 + km[v]]);
    asm volatile("" ::: "memory");

    o0 = wmma_f16(pa, vf0, o0);
    o1 = wmma_f16(pa, vf1, o1);
    o2 = wmma_f16(pa, vf2, o2);
    o3 = wmma_f16(pa, vf3, o3);
  }

  const int b = bh / NH, h = bh % NH;
  v8f os[4] = {o0, o1, o2, o3};
#pragma unroll
  for (int v = 0; v < 8; ++v) {
    const float inv = 1.0f / lrun[v];
    const int q = qt * 16 + v + 8 * hi;
#pragma unroll
    for (int g = 0; g < 4; ++g) {
      const int col = h * DH + g * 16 + ln;
      AO[((size_t)b * LL + q) * D_MODEL + col] = (_Float16)(os[g][v] * inv);
    }
  }
}

// ---------------------------------------------------------------- output projection
__global__ void oproj_gemm_kernel(const _Float16* __restrict__ ah,
                                  const _Float16* __restrict__ wh,
                                  const float*    __restrict__ bo,
                                  float* __restrict__ out) {
  const int lane = threadIdx.x & 31;
  const int wid  = blockIdx.x * (blockDim.x >> 5) + (threadIdx.x >> 5);
  const int ntiles = D_MODEL / 64;            // 12
  const int tm = wid / ntiles;
  const int tn = wid % ntiles;
  const int ln = lane & 15;
  const int hi = lane >> 4;

  int km[8];
#pragma unroll
  for (int v = 0; v < 8; ++v) km[v] = kmap(lane, v);

  const _Float16* arow = ah + (tm * 16 + ln) * D_MODEL;
  const _Float16* b0r  = wh + (tn * 64 +  0 + ln) * D_MODEL;
  const _Float16* b1r  = wh + (tn * 64 + 16 + ln) * D_MODEL;
  const _Float16* b2r  = wh + (tn * 64 + 32 + ln) * D_MODEL;
  const _Float16* b3r  = wh + (tn * 64 + 48 + ln) * D_MODEL;

  v8f acc0 = {}, acc1 = {}, acc2 = {}, acc3 = {};
  Frags fA, fB;
  load_frags(fA, arow, b0r, b1r, b2r, b3r, 0, km);
  for (int k0 = 0; k0 < D_MODEL; k0 += 64) {
    load_frags(fB, arow, b0r, b1r, b2r, b3r, k0 + 32, km);
    acc0 = wmma_f16(fA.a, fA.b0, acc0);
    acc1 = wmma_f16(fA.a, fA.b1, acc1);
    acc2 = wmma_f16(fA.a, fA.b2, acc2);
    acc3 = wmma_f16(fA.a, fA.b3, acc3);
    if (k0 + 64 < D_MODEL)
      load_frags(fA, arow, b0r, b1r, b2r, b3r, k0 + 64, km);
    acc0 = wmma_f16(fB.a, fB.b0, acc0);
    acc1 = wmma_f16(fB.a, fB.b1, acc1);
    acc2 = wmma_f16(fB.a, fB.b2, acc2);
    acc3 = wmma_f16(fB.a, fB.b3, acc3);
  }

  v8f accs[4] = {acc0, acc1, acc2, acc3};
#pragma unroll
  for (int g = 0; g < 4; ++g) {
#pragma unroll
    for (int v = 0; v < 8; ++v) {
      const int r = tm * 16 + v + 8 * hi;
      const int e = tn * 64 + g * 16 + ln;
      out[(size_t)r * D_MODEL + e] = accs[g][v] + bo[e];
    }
  }
}

// ---------------------------------------------------------------- launch
extern "C" void kernel_launch(void* const* d_in, const int* in_sizes, int n_in,
                              void* d_out, int out_size, void* d_ws, size_t ws_size,
                              hipStream_t stream) {
  const float* x    = (const float*)d_in[0];
  const float* Wqkv = (const float*)d_in[1];
  const float* bqkv = (const float*)d_in[2];
  const float* Wo   = (const float*)d_in[3];
  const float* bo   = (const float*)d_in[4];
  // d_in[5] = causal mask (implicit in kernel)

  _Float16* ws = (_Float16*)d_ws;
  size_t off = 0;
  _Float16* xh  = ws + off; off += (size_t)ML * D_MODEL;       // x in f16
  _Float16* wqh = ws + off; off += (size_t)E3 * D_MODEL;       // W_qkv f16
  _Float16* woh = ws + off; off += (size_t)D_MODEL * D_MODEL;  // W_o f16
  _Float16* Qh  = ws + off; off += (size_t)BB * NH * LL * DH;  // [b][h][l][d]
  _Float16* Kh  = ws + off; off += (size_t)BB * NH * LL * DH;  // [b][h][l][d]
  _Float16* Vt  = ws + off; off += (size_t)BB * NH * DH * LL;  // [b][h][d][l]
  _Float16* AO  = ws + off; off += (size_t)ML * D_MODEL;       // attn out f16

  f32_to_f16_kernel<<<512, 256, 0, stream>>>(x,    xh,  ML * D_MODEL);
  f32_to_f16_kernel<<<256, 256, 0, stream>>>(Wqkv, wqh, E3 * D_MODEL);
  f32_to_f16_kernel<<<64,  256, 0, stream>>>(Wo,   woh, D_MODEL * D_MODEL);

  // QKV GEMM: 512 M-tiles x 36 N-tiles, 1 wave each, 4 waves/block
  qkv_gemm_kernel<<<(512 * 36) / 4, 128, 0, stream>>>(xh, wqh, bqkv, Qh, Kh, Vt);

  // attention: 48 (b,h) x 128 q-tiles, 1 wave each
  attn_kernel<<<(48 * 128) / 4, 128, 0, stream>>>(Qh, Kh, Vt, AO);

  // output projection: 512 x 12 tiles
  oproj_gemm_kernel<<<(512 * 12) / 4, 128, 0, stream>>>(AO, woh, bo, (float*)d_out);
}
